// AnomalyAttention_48421461296003
// MI455X (gfx1250) — matrix operand order repaired and checked by reading, not compiled, not run on hardware
//
#include <hip/hip_runtime.h>
#include <hip/hip_bf16.h>
#include <math.h>

typedef __attribute__((ext_vector_type(16))) _Float16 v16h;
typedef __attribute__((ext_vector_type(8)))  float    v8f;
typedef __attribute__((ext_vector_type(4)))  float    v4f;

#define BB 4
#define LL 1024
#define HH 8
#define EE 64
#define DD 64
#define SSTR 1028   // LDS row stride (floats): 1028 % 8 == 4 -> disjoint bank halves for the two C row-groups

static __device__ __forceinline__ v8f wmma_f16(v16h a, v16h b, v8f c) {
    // v_wmma_f32_16x16x32_f16: (neg_a, A, neg_b, B, c_mod, C, reuse_a, reuse_b)
    return __builtin_amdgcn_wmma_f32_16x16x32_f16(false, a, false, b, (short)0, c, false, false);
}

// One workgroup = 4 wave32 = 128 threads, handles a 16-row tile of one (b,h).
// Phase 1: S = scale * mask(Q K^T)  (WMMA, f16 in / f32 acc) -> LDS
// Phase 2: row softmax -> series (global, NT streaming stores) + normalized probs kept in LDS
// Phase 3: V = P @ values (WMMA) -> V output (NT stores)
__global__ void __launch_bounds__(128)
anomaly_attn_kernel(const float* __restrict__ Q, const float* __restrict__ K,
                    const float* __restrict__ Vv,
                    float* __restrict__ Vout, float* __restrict__ series) {
    extern __shared__ float smem[];
    float* sS     = smem;                 // [16][SSTR]
    float* sInv   = smem + 16 * SSTR;     // [16] 1/rowsum

    const int ti0 = blockIdx.x;           // row-tile index 0..63
    const int h   = blockIdx.y;
    const int b   = blockIdx.z;
    const int i0  = ti0 * 16;

    const int lane = threadIdx.x & 31;
    const int wv   = threadIdx.x >> 5;    // wave id 0..3
    const int mn   = lane & 15;           // row (A/C) or col (B/C) within 16x16 tile
    const int hi   = lane >> 4;           // lane half

    const float scale = 0.125f;           // 1/sqrt(64)

    // ---------- Phase 1: scores ----------
    // A = Q tile rows i0..i0+15, contiguous k order (matches B): element i -> K = i + 16*hi
    v16h qa0, qa1;
    {
        const float* qrow = Q + (((size_t)b * LL + (i0 + mn)) * HH + h) * EE;
        #pragma unroll
        for (int i = 0; i < 16; ++i) {
            const int k = i + (hi << 4);
            qa0[i] = (_Float16)qrow[k];
            qa1[i] = (_Float16)qrow[k + 32];
        }
    }

    for (int tj = wv; tj <= ti0; tj += 4) {
        const int j0 = tj * 16;
        const float* krow = K + (((size_t)b * LL + (j0 + mn)) * HH + h) * EE + (hi << 4);
        if (tj + 4 <= ti0) {
            __builtin_prefetch(K + (((size_t)b * LL + (j0 + 64 + mn)) * HH + h) * EE, 0, 3);
        }
        // B = K^T tile, contiguous k order (matches A): element i -> Krow = i + 16*hi, col = mn
        v16h kb0, kb1;
        #pragma unroll
        for (int i = 0; i < 16; ++i) {
            kb0[i] = (_Float16)krow[i];
            kb1[i] = (_Float16)krow[i + 32];
        }
        v8f c = {};
        c = wmma_f16(qa0, kb0, c);
        c = wmma_f16(qa1, kb1, c);
        // C layout: VGPR v -> row M = v + 8*hi, col N = mn
        #pragma unroll
        for (int v = 0; v < 8; ++v) {
            const int row = v + (hi << 3);
            const int jg  = j0 + mn;
            const int ig  = i0 + row;
            sS[row * SSTR + jg] = (jg > ig) ? -1e30f : c[v] * scale;
        }
    }
    __syncthreads();

    // ---------- Phase 2: softmax ----------
    const int limit = (ti0 + 1) * 16;     // valid score columns [0, limit)
    {
        const int r  = threadIdx.x >> 3;  // row 0..15
        const int t8 = threadIdx.x & 7;   // 8 threads/row
        float mx = -1e30f;
        for (int j = t8; j < limit; j += 8) mx = fmaxf(mx, sS[r * SSTR + j]);
        #pragma unroll
        for (int o = 4; o >= 1; o >>= 1) mx = fmaxf(mx, __shfl_xor(mx, o, 8));
        float sum = 0.0f;
        for (int j = t8; j < LL; j += 8) {
            float e = 0.0f;
            if (j < limit) { e = __expf(sS[r * SSTR + j] - mx); sum += e; }
            sS[r * SSTR + j] = e;
        }
        #pragma unroll
        for (int o = 4; o >= 1; o >>= 1) sum += __shfl_xor(sum, o, 8);
        if (t8 == 0) sInv[r] = 1.0f / sum;
    }
    __syncthreads();

    // normalize in LDS + stream series out with b128 NT stores (write-once data:
    // keep L2 free for K/V tile reuse across the 64 row-tile workgroups per (b,h))
    {
        float* serBase = series + (((size_t)b * HH + h) * LL + i0) * (size_t)LL;
        for (int r = 0; r < 16; ++r) {
            const float inv = sInv[r];
            #pragma unroll
            for (int j4 = (int)threadIdx.x * 4; j4 < LL; j4 += 512) {
                v4f p;
                #pragma unroll
                for (int c = 0; c < 4; ++c) {
                    const float pv = sS[r * SSTR + j4 + c] * inv;
                    sS[r * SSTR + j4 + c] = pv;
                    p[c] = pv;
                }
                __builtin_nontemporal_store(p, (v4f*)(serBase + (size_t)r * LL + j4));
            }
        }
    }
    __syncthreads();

    // ---------- Phase 3: V = P @ values ----------
    {
        const int n0 = wv * 16;                    // each wave owns 16 cols of D=64
        const int ksteps = (ti0 + 2) >> 1;         // ceil(limit/32)
        v8f acc = {};
        for (int ks = 0; ks < ksteps; ++ks) {
            const int kb = ks * 32;
            v16h pa, vb;
            #pragma unroll
            for (int i = 0; i < 16; ++i) {         // A = probs, same 16-bit A layout
                const int k = i + (hi << 4);
                pa[i] = (_Float16)sS[mn * SSTR + kb + k];
            }
            const float* vcol = Vv + (((size_t)b * LL + (kb + (hi << 4))) * HH + h) * DD + n0 + mn;
            #pragma unroll
            for (int i = 0; i < 16; ++i) {         // B = values, Krow = i + 16*hi
                vb[i] = (_Float16)vcol[(size_t)i * (HH * DD)];
            }
            acc = wmma_f16(pa, vb, acc);
        }
        #pragma unroll
        for (int v = 0; v < 8; ++v) {
            const int row = v + (hi << 3);
            __builtin_nontemporal_store(
                acc[v],
                Vout + (((size_t)b * LL + (i0 + row)) * HH + h) * DD + n0 + mn);
        }
    }
}

// One block per (b,h,i) row: compute sg once, stream prior + sigma_out.
// 256 threads x float4 = exactly 1024 columns; all stores are b128 NT.
__global__ void __launch_bounds__(256)
anomaly_prior_kernel(const float* __restrict__ sigma,
                     float* __restrict__ prior, float* __restrict__ sigma_out) {
    const int i = blockIdx.x, h = blockIdx.y, b = blockIdx.z;
    const float sv = sigma[((size_t)b * LL + i) * HH + h];
    const float s  = 1.0f / (1.0f + __expf(-5.0f * sv)) + 1e-5f;
    const float sg = exp2f(s * 1.5849625007211562f) - 1.0f;  // 3^s - 1
    const float inv2 = 1.0f / (2.0f * sg * sg);
    const float coef = 0.3989422804014327f / sg;             // 1/(sqrt(2pi)*sg)
    const size_t base = (((size_t)b * HH + h) * LL + i) * (size_t)LL;

    const int j4 = (int)threadIdx.x * 4;
    v4f pr, sg4;
    #pragma unroll
    for (int c = 0; c < 4; ++c) {
        const float d = (float)(i - (j4 + c));
        pr[c]  = coef * __expf(-(d * d) * inv2);
        sg4[c] = sg;
    }
    __builtin_nontemporal_store(pr,  (v4f*)(prior + base + j4));
    __builtin_nontemporal_store(sg4, (v4f*)(sigma_out + base + j4));
}

extern "C" void kernel_launch(void* const* d_in, const int* in_sizes, int n_in,
                              void* d_out, int out_size, void* d_ws, size_t ws_size,
                              hipStream_t stream) {
    (void)in_sizes; (void)n_in; (void)d_ws; (void)ws_size; (void)out_size;
    const float* Q  = (const float*)d_in[0];
    const float* K  = (const float*)d_in[1];
    const float* Vv = (const float*)d_in[2];
    const float* Sg = (const float*)d_in[3];

    float* out = (float*)d_out;
    const size_t V_OFF   = 0;
    const size_t SER_OFF = (size_t)BB * LL * HH * DD;                       //  2,097,152
    const size_t PRI_OFF = SER_OFF + (size_t)BB * HH * LL * LL;             // 35,651,584
    const size_t SIG_OFF = PRI_OFF + (size_t)BB * HH * LL * LL;             // 69,206,016

    const size_t shmem = (16 * SSTR + 16) * sizeof(float);                  // ~65.9 KB (WGP has 320 KB LDS)

    dim3 gridA(LL / 16, HH, BB);
    anomaly_attn_kernel<<<gridA, 128, shmem, stream>>>(Q, K, Vv,
                                                       out + V_OFF, out + SER_OFF);

    dim3 gridP(LL, HH, BB);
    anomaly_prior_kernel<<<gridP, 256, 0, stream>>>(Sg, out + PRI_OFF, out + SIG_OFF);
}
